// FeatureMatching_81140522156193
// MI455X (gfx1250) — matrix operand order repaired and checked by reading, not compiled, not run on hardware
//
#include <hip/hip_runtime.h>
#include <hip/hip_bf16.h>
#include <math.h>
#include <stdint.h>

typedef __attribute__((ext_vector_type(16))) _Float16 v16h;
typedef __attribute__((ext_vector_type(8)))  float    v8f;
typedef _Float16 half_t;

#define HQ 288
#define WQ 288
#define HWQ (288*288)
#define NPATCH 9216   // 96*96
#define KPAD 160      // 144 padded to 5 K-tiles of 32

__device__ __forceinline__ v8f wmma16(v16h a, v16h b, v8f c) {
  // D = A(16x32 f16) * B(32x16 f16) + C(16x16 f32)
  return __builtin_amdgcn_wmma_f32_16x16x32_f16(
      /*neg_a=*/false, a, /*neg_b=*/false, b,
      /*c_mod=*/(short)0, c, /*reuse_a=*/false, /*reuse_b=*/false);
}

// ---------------------------------------------------------------------------
// Prep kernels
// ---------------------------------------------------------------------------

// Generic f32 -> f16 convert (weights w2/w3/w4, and raw input images).
__global__ void cvt_f32_f16(const float* __restrict__ s, half_t* __restrict__ d, int n) {
  int i = blockIdx.x * blockDim.x + threadIdx.x;
  if (i < n) d[i] = (half_t)s[i];
}

// Fold MeanShift ((x-mean)/std) into conv1: w' = w/std[ci], b' = b - sum w*mean/std.
// wp1 rows padded to 64 (K-tiles of 32), zeros for k >= 54.
__global__ void fold_w1_kernel(const float* __restrict__ w1, const float* __restrict__ b1,
                               half_t* __restrict__ wp1, float* __restrict__ bf1) {
  const float VMEAN[6] = {0.485f, 0.456f, 0.406f, 0.5f, 0.5f, 0.5f};
  const float VSTD[6]  = {0.229f, 0.224f, 0.225f, 0.1f, 0.1f, 0.1f};
  int n = blockIdx.x * blockDim.x + threadIdx.x;
  if (n >= 64) return;
  float bacc = b1[n];
  for (int k = 0; k < 64; ++k) {
    half_t v = (half_t)0.f;
    if (k < 54) {
      int ci = k / 9;
      float wv = w1[n * 54 + k];
      v = (half_t)(wv / VSTD[ci]);
      bacc -= wv * VMEAN[ci] / VSTD[ci];
    }
    wp1[n * 64 + k] = v;
  }
  bf1[n] = bacc;
}

// ---------------------------------------------------------------------------
// Implicit-GEMM WMMA convolution.
//   M = 16 consecutive pixels in one row, N = NT*16 output channels per wave,
//   K = cin*KS*KS (k = ci*KS*KS + kh*KS + kw, matching OIHW weight flattening).
// in : [cin][H][W] f16     wt : [cout][ktiles*32] f16 (zero-padded rows)
// out: [cout][H][W] f16    act: 0 = ReLU, 1 = LeakyReLU(0.2)
// ---------------------------------------------------------------------------
template <int KS, int NT>
__global__ __launch_bounds__(256)
void conv_wmma(const half_t* __restrict__ in, const half_t* __restrict__ wt,
               const float* __restrict__ bias, half_t* __restrict__ out,
               int kvalid, int ktiles, int H, int W, int act) {
  const int lane   = threadIdx.x & 31;
  const int widblk = threadIdx.x >> 5;
  const int wtiles = W >> 4;                 // 18
  const int mt     = blockIdx.x * 8 + widblk;
  if (mt >= H * wtiles) return;              // wave-uniform (exact grids anyway)
  const int y    = mt / wtiles;
  const int x0   = (mt - y * wtiles) << 4;
  const int hi   = lane >> 4;
  const int m    = lane & 15;                // A row (pixel) & B column (channel)
  const int xpix = x0 + m;
  const int nb   = blockIdx.y * (NT * 16);
  const int kstride = ktiles * 32;

  v8f acc[NT];
#pragma unroll
  for (int nt = 0; nt < NT; ++nt) acc[nt] = (v8f){0.f,0.f,0.f,0.f,0.f,0.f,0.f,0.f};

  for (int kt = 0; kt < ktiles; ++kt) {
    // ---- A fragment: a[i] <- A[m][ kt*32 + 16*hi + i ]
    v16h a;
#pragma unroll
    for (int i = 0; i < 16; ++i) {
      int k = kt * 32 + hi * 16 + i;
      half_t val = (half_t)0.f;
      if (k < kvalid) {
        if (KS == 3) {
          int ci = k / 9;
          int fp = k - ci * 9;
          int kh = fp / 3;
          int kw = fp - kh * 3;
          int iy = y + kh - 1;
          int ix = xpix + kw - 1;
          if (iy >= 0 && iy < H && ix >= 0 && ix < W)
            val = in[((size_t)ci * H + iy) * W + ix];
        } else {  // 1x1
          val = in[(size_t)k * H * W + (size_t)y * W + xpix];
        }
      }
      a[i] = val;
    }
    // ---- B fragments (contiguous 32B weight rows) + WMMA
#pragma unroll
    for (int nt = 0; nt < NT; ++nt) {
      int n = nb + nt * 16 + m;
      v16h bfrag = *(const v16h*)(wt + (size_t)n * kstride + kt * 32 + hi * 16);
      acc[nt] = wmma16(a, bfrag, acc[nt]);
    }
  }

  // ---- Epilogue: bias + activation, pack 8 f16 -> one 16B store per lane.
#pragma unroll
  for (int nt = 0; nt < NT; ++nt) {
    int n = nb + nt * 16 + m;
    float bv = bias[n];
    union { uint4 u; half_t h[8]; } pk;
#pragma unroll
    for (int v = 0; v < 8; ++v) {
      float f = acc[nt][v] + bv;
      f = (act == 0) ? fmaxf(f, 0.0f) : (f > 0.0f ? f : 0.2f * f);
      pk.h[v] = (half_t)f;
    }
    half_t* op = out + ((size_t)n * H + y) * W + x0 + hi * 8;  // 16B aligned
    *(uint4*)op = pk.u;
  }
}

// ---------------------------------------------------------------------------
// Unfold(3x3, stride 3) + L2-normalize -> dst[patch][160] f16 (zeros pad 144..159)
// ---------------------------------------------------------------------------
__global__ void patch_kernel(const half_t* __restrict__ f, half_t* __restrict__ dst) {
  int t = blockIdx.x * blockDim.x + threadIdx.x;
  if (t >= NPATCH) return;
  int qy = t / 96, qx = t - qy * 96;
  float ss = 0.f;
  for (int c = 0; c < 16; ++c)
    for (int kh = 0; kh < 3; ++kh)
      for (int kw = 0; kw < 3; ++kw) {
        float v = (float)f[((size_t)c * HQ + (qy * 3 + kh)) * WQ + (qx * 3 + kw)];
        ss += v * v;
      }
  float inv = 1.0f / fmaxf(sqrtf(ss), 1e-12f);
  half_t* o = dst + (size_t)t * KPAD;
  int d = 0;
  for (int c = 0; c < 16; ++c)
    for (int kh = 0; kh < 3; ++kh)
      for (int kw = 0; kw < 3; ++kw) {
        float v = (float)f[((size_t)c * HQ + (qy * 3 + kh)) * WQ + (qx * 3 + kw)];
        o[d++] = (half_t)(v * inv);
      }
  for (int z = 144; z < KPAD; ++z) o[z] = (half_t)0.f;
}

// ---------------------------------------------------------------------------
// Correlation + fused row max/argmax.
//   y[l][m] = kp[l] . qp[m]; reduce over l.  A = kp rows, staged into LDS by
//   wave 0 via CDNA5 async copy (GLOBAL_LOAD_ASYNC_TO_LDS_B128, ASYNCcnt),
//   shared by all 8 waves.  B = qp columns (register-resident, 5 K-tiles).
//   D layout: lane=col, VGPR v = row v + 8*hi -> in-register running max,
//   shfl_xor(16) merge with first-index tie-break.
// ---------------------------------------------------------------------------
__global__ __launch_bounds__(256)
void corr_kernel(const half_t* __restrict__ qp, const half_t* __restrict__ kp,
                 float* __restrict__ outRel, float* __restrict__ outIdx) {
  const int lane = threadIdx.x & 31;
  const int w    = threadIdx.x >> 5;
  const int b    = blockIdx.y;
  const half_t* qpB = qp + (size_t)b * NPATCH * KPAD;
  const half_t* kpB = kp + (size_t)b * NPATCH * KPAD;
  const int cbase = blockIdx.x * 128 + w * 16;
  const int hi = lane >> 4, n = lane & 15;

  // Register-resident B (query columns), 5 K-tiles.
  v16h bf[5];
#pragma unroll
  for (int kt = 0; kt < 5; ++kt)
    bf[kt] = *(const v16h*)(qpB + (size_t)(cbase + n) * KPAD + kt * 32 + hi * 16);

  __shared__ __align__(16) half_t As[16 * KPAD];  // 5120 B key-row tile

  // LDS-relative byte address of As: flat LDS pointers carry the wave-relative
  // LDS address in addr[31:0] (ISA 10.2 aperture rules).
  const uint32_t ldsbase = (uint32_t)(uintptr_t)(void*)&As[0];

  float bv = -3.4e38f;
  int   bi = 0;

  for (int lb = 0; lb < NPATCH; lb += 16) {
    __syncthreads();  // previous tile fully consumed before overwrite
    if (w == 0) {
      // Async DMA: 10 x (32 lanes x 16B) = 5120 B straight into LDS, no VGPR
      // round trip.  Tracked by ASYNCcnt.
      const char* gsrc = (const char*)(kpB + (size_t)lb * KPAD);
#pragma unroll
      for (int j = 0; j < 10; ++j) {
        uint32_t loff = ldsbase + (uint32_t)(j * 512 + lane * 16);
        uint64_t ga   = (uint64_t)(uintptr_t)(gsrc + j * 512 + lane * 16);
        asm volatile("global_load_async_to_lds_b128 %0, %1, off"
                     :: "v"(loff), "v"(ga) : "memory");
      }
      asm volatile("s_wait_asynccnt 0x0" ::: "memory");
    }
    __syncthreads();  // tile visible to all waves

    v8f acc = (v8f){0.f,0.f,0.f,0.f,0.f,0.f,0.f,0.f};
#pragma unroll
    for (int kt = 0; kt < 5; ++kt) {
      union { uint4 u[2]; v16h v; } au;
      const half_t* ar = As + n * KPAD + kt * 32 + hi * 16;
      au.u[0] = *(const uint4*)ar;         // k = kt*32 + 16*hi + 0..7
      au.u[1] = *(const uint4*)(ar + 8);   // k = kt*32 + 16*hi + 8..15
      acc = wmma16(au.v, bf[kt], acc);
    }
#pragma unroll
    for (int v = 0; v < 8; ++v) {
      float val = acc[v];
      int row = lb + v + hi * 8;
      if (val > bv) { bv = val; bi = row; }  // strictly-greater => first argmax
    }
  }

  // Merge the two half-wave partials for the same column (rows 0-7 vs 8-15 sets).
  float obv = __shfl_xor(bv, 16, 32);
  int   obi = __shfl_xor(bi, 16, 32);
  if (obv > bv || (obv == bv && obi < bi)) { bv = obv; bi = obi; }

  if (lane < 16) {
    int col = cbase + n;
    outRel[(size_t)b * NPATCH + col] = bv;
    outIdx[(size_t)b * NPATCH + col] = (float)bi;
  }
}

// ---------------------------------------------------------------------------
// Host-side orchestration
// ---------------------------------------------------------------------------
extern "C" void kernel_launch(void* const* d_in, const int* in_sizes, int n_in,
                              void* d_out, int out_size, void* d_ws, size_t ws_size,
                              hipStream_t stream) {
  (void)in_sizes; (void)n_in; (void)out_size; (void)ws_size;
  const float* query = (const float*)d_in[0];
  const float* keyp  = (const float*)d_in[1];
  // d_in[2] = flag_8k (static 1) -> unused
  const float* w1 = (const float*)d_in[3];
  const float* b1 = (const float*)d_in[4];
  const float* w2 = (const float*)d_in[5];
  const float* b2 = (const float*)d_in[6];
  const float* w3 = (const float*)d_in[7];
  const float* b3 = (const float*)d_in[8];
  const float* w4 = (const float*)d_in[9];
  const float* b4 = (const float*)d_in[10];

  char* ws = (char*)d_ws;
  auto alloc = [&](size_t bytes) -> char* {
    char* p = ws;
    ws += (bytes + 255) & ~(size_t)255;
    return p;
  };
  half_t* wp1  = (half_t*)alloc((size_t)64 * 64 * 2);
  half_t* wp2  = (half_t*)alloc((size_t)64 * 576 * 2);
  half_t* wp3  = (half_t*)alloc((size_t)128 * 576 * 2);
  half_t* wp4  = (half_t*)alloc((size_t)16 * 128 * 2);
  float*  bf1  = (float*) alloc((size_t)64 * 4);
  half_t* xin  = (half_t*)alloc((size_t)6   * HWQ * 2);
  half_t* bufA = (half_t*)alloc((size_t)64  * HWQ * 2);
  half_t* bufB = (half_t*)alloc((size_t)64  * HWQ * 2);
  half_t* bufC = (half_t*)alloc((size_t)128 * HWQ * 2);
  half_t* bufF = (half_t*)alloc((size_t)16  * HWQ * 2);
  half_t* qpb  = (half_t*)alloc((size_t)2 * NPATCH * KPAD * 2);
  half_t* kpb  = (half_t*)alloc((size_t)2 * NPATCH * KPAD * 2);

  // Weight prep (once).
  fold_w1_kernel<<<1, 64, 0, stream>>>(w1, b1, wp1, bf1);
  cvt_f32_f16<<<(64 * 576 + 255) / 256, 256, 0, stream>>>(w2, wp2, 64 * 576);
  cvt_f32_f16<<<(128 * 576 + 255) / 256, 256, 0, stream>>>(w3, wp3, 128 * 576);
  cvt_f32_f16<<<(16 * 128 + 255) / 256, 256, 0, stream>>>(w4, wp4, 16 * 128);

  // Shared encoder, one image at a time through reused buffers.
  for (int img = 0; img < 4; ++img) {
    const float* src = (img < 2) ? query + (size_t)img * 6 * HWQ
                                 : keyp  + (size_t)(img - 2) * 6 * HWQ;
    int nin = 6 * HWQ;
    cvt_f32_f16<<<(nin + 255) / 256, 256, 0, stream>>>(src, xin, nin);
    // conv1: 6->64, K=54 (padded to 64), MeanShift folded, ReLU
    conv_wmma<3, 4><<<dim3(648, 1), 256, 0, stream>>>(xin,  wp1, bf1, bufA, 54, 2, HQ, WQ, 0);
    // conv2: 64->64, K=576, ReLU
    conv_wmma<3, 4><<<dim3(648, 1), 256, 0, stream>>>(bufA, wp2, b2,  bufB, 576, 18, HQ, WQ, 0);
    // conv3: 64->128, K=576, ReLU (grid.y = 2 channel groups of 64)
    conv_wmma<3, 4><<<dim3(648, 2), 256, 0, stream>>>(bufB, wp3, b3,  bufC, 576, 18, HQ, WQ, 0);
    // conv4: 1x1 128->16, K=128, LeakyReLU(0.2)
    conv_wmma<1, 1><<<dim3(648, 1), 256, 0, stream>>>(bufC, wp4, b4,  bufF, 128, 4, HQ, WQ, 1);
    // unfold + l2norm -> padded patch rows
    half_t* pdst = (img < 2) ? qpb + (size_t)img * NPATCH * KPAD
                             : kpb + (size_t)(img - 2) * NPATCH * KPAD;
    patch_kernel<<<NPATCH / 256, 256, 0, stream>>>(bufF, pdst);
  }

  // Correlation + max/argmax. Output: [2][9216] relevance then [2][9216] indices.
  float* outRel = (float*)d_out;
  float* outIdx = outRel + 2 * NPATCH;
  corr_kernel<<<dim3(NPATCH / 128, 2), 256, 0, stream>>>(qpb, kpb, outRel, outIdx);
}
